// TinyTransformerBlock_14577119002934
// MI455X (gfx1250) — matrix operand order repaired and checked by reading, not compiled, not run on hardware
//
#include <hip/hip_runtime.h>
#include <hip/hip_bf16.h>

// ---------------------------------------------------------------------------
// TinyTransformerBlock for MI455X (gfx1250, wave32)
//   B=4, S=2048, D=1024, DFF=4096
// GEMMs: V_WMMA_F32_16X16X32_BF16, tiles staged by the Tensor Data Mover
// (TENSOR_LOAD_TO_LDS, TENSORcnt) with double-buffered LDS.
// ---------------------------------------------------------------------------

typedef __bf16 bf;
typedef __attribute__((ext_vector_type(16))) __bf16 v16bf;
typedef __attribute__((ext_vector_type(8)))  __bf16 v8bf;
typedef __attribute__((ext_vector_type(8)))  float  v8f;
typedef __attribute__((ext_vector_type(4)))  unsigned int u32x4;
typedef __attribute__((ext_vector_type(8)))  int i32x8;
typedef __attribute__((ext_vector_type(4)))  int i32x4;

union Frag {
    v16bf v;
    struct { v8bf lo; v8bf hi; } h;
};

#define TM 128          // workgroup tile M (4 waves along M)
#define TN 128          // workgroup tile N (2 waves along N, 64 each)
#define TK 32           // K step per WMMA
#define LDSS 40         // LDS row stride in halves: 32 data + 8 pad (16B align kept)
#define BUFE (TM * LDSS) // elements per LDS buffer

// LDS byte offset of a generic pointer that points into shared memory.
// (evaluated at runtime inside the kernel -> plain ptrtoint, no static init)
typedef __attribute__((address_space(3))) char as3char;
__device__ __forceinline__ unsigned lds_byte_off(void* p) {
    return (unsigned)(unsigned long long)(as3char*)p;
}

// One TDM 2D tile load: rows x 32 bf16, row stride = rowStrideElems, into LDS
// at byte offset ldsOff with a 16-byte pad after every 64-byte row (-> LDSS=40).
// D# layout per CDNA5 ISA.
__device__ __forceinline__ void tdm_load_tile(unsigned ldsOff,
                                              unsigned long long gaddr,
                                              unsigned rows,
                                              long long rowStrideElems)
{
    u32x4 g0;
    g0[0] = 1u;                                          // count=1 (valid user D#)
    g0[1] = ldsOff;                                      // lds_addr (bytes)
    g0[2] = (unsigned)(gaddr & 0xFFFFFFFFull);           // global_addr[31:0]
    g0[3] = (unsigned)((gaddr >> 32) & 0x1FFFFFFull)     // global_addr[56:32]
          | (2u << 30);                                  // type = 2 ("image")
    const unsigned dim0 = (unsigned)TK;                  // tile/tensor width (elems)
    const unsigned dim1 = rows;
    const unsigned long long s0 = (unsigned long long)rowStrideElems;
    i32x8 g1;
    g1[0] = (int)((1u << 16)      // data_size = 1 -> 2 bytes
                | (1u << 20)      // pad_enable
                | (3u << 22)      // pad_interval code 3 -> every 16 DWORDs (64 B row)
                | (3u << 25));    // pad_amount  code 3 -> 4 DWORDs (16 B) pad
    g1[1] = (int)((dim0 & 0xFFFFu) << 16);                       // tensor_dim0[15:0]
    g1[2] = (int)(((dim0 >> 16) & 0xFFFFu) | ((dim1 & 0xFFFFu) << 16));
    g1[3] = (int)(((dim1 >> 16) & 0xFFFFu) | (dim0 << 16));      // tile_dim0 = 32
    g1[4] = (int)(dim1 & 0xFFFFu);                               // tile_dim1 = rows, tile_dim2=0
    g1[5] = (int)(unsigned)(s0 & 0xFFFFFFFFull);                 // tensor_dim0_stride lo
    g1[6] = (int)(unsigned)((s0 >> 32) & 0xFFFFull);             // stride hi, dim1_stride lo=0
    g1[7] = 0;
    const i32x4 z4 = {0, 0, 0, 0};
    const i32x8 z8 = {0, 0, 0, 0, 0, 0, 0, 0};
    // 6-arg form on this toolchain: (g0, g1, g2, g3, g4, cpol)
    __builtin_amdgcn_tensor_load_to_lds(g0, g1, z4, z4, z8, 0);
}

// C[M x N] = act( A[M x K] * BT[N x K]^T )  (+ optional fp32 residual)
// ACT: 0 none, 1 relu.  OUT_F32: write fp32, else bf16.  ADD_RES: += res.
template <int ACT, int OUT_F32, int ADD_RES>
__global__ __launch_bounds__(256) void wmma_gemm(
    const bf* __restrict__ A, const bf* __restrict__ BT,
    void* __restrict__ Cv, const float* __restrict__ res,
    int M, int N, int K,
    long long sAb, long long sBb, long long sCb)
{
    // double-buffered: buf0 A | buf1 A | buf0 B | buf1 B  -> 40 KB total
    __shared__ bf smem[4 * BUFE];

    const int tid = threadIdx.x;
    const int z   = blockIdx.z;
    A  += (long long)z * sAb;
    BT += (long long)z * sBb;
    const long long cbase = (long long)z * sCb;

    const int bm = blockIdx.y * TM;
    const int bn = blockIdx.x * TN;

    const int wave = tid >> 5;           // 8 waves (wave32)
    const int lane = tid & 31;
    const int m0   = (wave >> 1) * 32;   // 4 waves along M, 32 rows each
    const int n0   = (wave & 1) * 64;    // 2 waves along N, 64 cols each
    const int lrow = lane & 15;
    const int kbA  = (lane >> 4) * 8;    // A frag: K in {kb..kb+7, kb+16..kb+23}
    const int kbB  = (lane >> 4) * 8;    // B frag: K in {kb..kb+7, kb+16..kb+23}

    const v8f vzero = {0.f, 0.f, 0.f, 0.f, 0.f, 0.f, 0.f, 0.f};
    v8f acc[2][4];
#pragma unroll
    for (int i = 0; i < 2; ++i)
#pragma unroll
        for (int j = 0; j < 4; ++j) acc[i][j] = vzero;

    const unsigned smemOff = lds_byte_off(smem);      // runtime ptrtoint (no static init)
    const unsigned long long gA =
        (unsigned long long)(const void*)A + (unsigned long long)((long long)bm * K) * 2ull;
    const unsigned long long gB =
        (unsigned long long)(const void*)BT + (unsigned long long)((long long)bn * K) * 2ull;

    const int nK = K / TK;

    // ---- TDM prologue: wave 0 pulls tile 0 (A+B) into buffer 0 -------------
    if (wave == 0) {
        tdm_load_tile(smemOff + 0u * (BUFE * 2u), gA, TM, K);
        tdm_load_tile(smemOff + 2u * (BUFE * 2u), gB, TN, K);
    }

    for (int k = 0; k < nK; ++k) {
        const int cur = k & 1;
        if (wave == 0) {
            if (k + 1 < nK) {
                const unsigned nxt = (unsigned)(cur ^ 1);
                const unsigned long long koff = (unsigned long long)(k + 1) * TK * 2ull;
                tdm_load_tile(smemOff + nxt * (BUFE * 2u),        gA + koff, TM, K);
                tdm_load_tile(smemOff + (2u + nxt) * (BUFE * 2u), gB + koff, TN, K);
                __builtin_amdgcn_s_wait_tensorcnt(2);  // only the 2 just issued in flight
            } else {
                __builtin_amdgcn_s_wait_tensorcnt(0);
            }
        }
        __syncthreads();   // buffer `cur` is ready for everyone

        bf* sAc = smem + cur * BUFE;          // computed, not a const-init pointer array
        bf* sBc = smem + (2 + cur) * BUFE;

        Frag a[2], b[4];
#pragma unroll
        for (int i = 0; i < 2; ++i) {
            const bf* p = &sAc[(m0 + i * 16 + lrow) * LDSS + kbA];
            a[i].h.lo = *(const v8bf*)p;            // K = kb..kb+7
            a[i].h.hi = *(const v8bf*)(p + 16);     // K = kb+16..kb+23
        }
#pragma unroll
        for (int j = 0; j < 4; ++j) {
            const bf* p = &sBc[(n0 + j * 16 + lrow) * LDSS + kbB];
            b[j].h.lo = *(const v8bf*)p;
            b[j].h.hi = *(const v8bf*)(p + 16);
        }
#pragma unroll
        for (int i = 0; i < 2; ++i)
#pragma unroll
            for (int j = 0; j < 4; ++j)
                acc[i][j] = __builtin_amdgcn_wmma_f32_16x16x32_bf16(
                    false, a[i].v, false, b[j].v,
                    (short)0, acc[i][j], false, false);

        __syncthreads();   // all waves done with buffer `cur` before it is reloaded
    }

    // ---- store: VGPR r, lanes 0-15 -> M=r, lanes 16-31 -> M=8+r; N = lane%16
    const int rh = (lane >> 4) * 8;
#pragma unroll
    for (int i = 0; i < 2; ++i)
#pragma unroll
        for (int j = 0; j < 4; ++j)
#pragma unroll
            for (int r = 0; r < 8; ++r) {
                int gm = bm + m0 + i * 16 + rh + r;
                int gn = bn + n0 + j * 16 + lrow;
                float v = acc[i][j][r];
                if (ACT) v = v > 0.f ? v : 0.f;
                long long idx = cbase + (long long)gm * N + gn;
                if (OUT_F32) {
                    float o = v;
                    if (ADD_RES) o += res[idx];
                    ((float*)Cv)[idx] = o;
                } else {
                    ((bf*)Cv)[idx] = (bf)v;
                }
            }
}

// one block per row of 2048 fp32 scores -> bf16 probabilities
__global__ __launch_bounds__(256) void softmax_rows(
    const float* __restrict__ Sc, bf* __restrict__ P, int n)
{
    const long long row = blockIdx.x;
    const float* s = Sc + row * n;
    bf* p = P + row * n;
    float vals[8];
    float m = -3.0e38f;
#pragma unroll
    for (int i = 0; i < 8; ++i) {
        vals[i] = s[threadIdx.x + i * 256];
        m = fmaxf(m, vals[i]);
    }
    __shared__ float red[256];
    red[threadIdx.x] = m; __syncthreads();
    for (int off = 128; off > 0; off >>= 1) {
        if (threadIdx.x < off)
            red[threadIdx.x] = fmaxf(red[threadIdx.x], red[threadIdx.x + off]);
        __syncthreads();
    }
    m = red[0]; __syncthreads();
    float sum = 0.f;
#pragma unroll
    for (int i = 0; i < 8; ++i) { vals[i] = __expf(vals[i] - m); sum += vals[i]; }
    red[threadIdx.x] = sum; __syncthreads();
    for (int off = 128; off > 0; off >>= 1) {
        if (threadIdx.x < off)
            red[threadIdx.x] += red[threadIdx.x + off];
        __syncthreads();
    }
    float inv = 1.f / red[0];
#pragma unroll
    for (int i = 0; i < 8; ++i)
        p[threadIdx.x + i * 256] = (bf)(vals[i] * inv);
}

// bf16 [R x C] -> [C x R] per batch (grid.z), batch stride R*C
__global__ __launch_bounds__(256) void transpose_bf16(
    const bf* __restrict__ in, bf* __restrict__ out, int R, int C)
{
    __shared__ bf t[32][33];
    long long boff = (long long)blockIdx.z * R * C;
    int c0 = blockIdx.x * 32, r0 = blockIdx.y * 32;
    int tx = threadIdx.x & 31, ty = threadIdx.x >> 5;
    for (int rr = ty; rr < 32; rr += 8)
        t[rr][tx] = in[boff + (long long)(r0 + rr) * C + c0 + tx];
    __syncthreads();
    for (int rr = ty; rr < 32; rr += 8)
        out[boff + (long long)(c0 + rr) * R + r0 + tx] = t[tx][rr];
}

// fp32 [R x C] -> bf16 [C x R]   (weight transpose+convert)
__global__ __launch_bounds__(256) void transpose_cvt(
    const float* __restrict__ in, bf* __restrict__ out, int R, int C)
{
    __shared__ float t[32][33];
    int c0 = blockIdx.x * 32, r0 = blockIdx.y * 32;
    int tx = threadIdx.x & 31, ty = threadIdx.x >> 5;
    for (int rr = ty; rr < 32; rr += 8)
        t[rr][tx] = in[(long long)(r0 + rr) * C + c0 + tx];
    __syncthreads();
    for (int rr = ty; rr < 32; rr += 8)
        out[(long long)(c0 + rr) * R + r0 + tx] = (bf)t[tx][rr];
}

__global__ __launch_bounds__(256) void cvt_bf16(
    const float* __restrict__ in, bf* __restrict__ out, long long n)
{
    long long i = (long long)blockIdx.x * 256 + threadIdx.x;
    if (i < n) out[i] = (bf)in[i];
}

extern "C" void kernel_launch(void* const* d_in, const int* in_sizes, int n_in,
                              void* d_out, int out_size, void* d_ws, size_t ws_size,
                              hipStream_t stream)
{
    (void)in_sizes; (void)n_in; (void)out_size; (void)ws_size;
    const int B = 4, S = 2048, D = 1024, DFF = 4096;
    const float* x  = (const float*)d_in[0];
    const float* wq = (const float*)d_in[1];
    const float* wk = (const float*)d_in[2];
    const float* wv = (const float*)d_in[3];
    const float* wo = (const float*)d_in[4];
    const float* w1 = (const float*)d_in[5];
    const float* w2 = (const float*)d_in[6];

    // ---- workspace layout (~176 MB), hand-aliased by liveness --------------
    char* ws = (char*)d_ws;
    size_t off = 0;
    auto alloc = [&](size_t bytes) -> char* {
        char* p = ws + off;
        off += (bytes + 255) & ~(size_t)255;
        return p;
    };
    bf* wqT = (bf*)alloc((size_t)D * D   * 2);
    bf* wkT = (bf*)alloc((size_t)D * D   * 2);
    bf* wvT = (bf*)alloc((size_t)D * D   * 2);
    bf* woT = (bf*)alloc((size_t)D * D   * 2);
    bf* w1T = (bf*)alloc((size_t)D * DFF * 2);   // [DFF][D]
    bf* w2T = (bf*)alloc((size_t)DFF * D * 2);   // [D][DFF]
    bf* Q   = (bf*)alloc((size_t)B * S * D * 2);
    bf* Kb  = (bf*)alloc((size_t)B * S * D * 2);
    bf* Vt  = (bf*)alloc((size_t)B * S * D * 2); // [B][D][S]
    bf* xb  = (bf*)alloc((size_t)B * S * D * 2); // exactly 16 MiB (256B multiple)
    bf* V   = (bf*)alloc((size_t)B * S * D * 2); // contiguous after xb
    char* big = alloc((size_t)B * S * S * 4);    // scores fp32, later hidden bf16
    float* scores = (float*)big;                 // live: phase 3-4
    bf*    hidden = (bf*)big;                    // live: phase 7-8
    bf*    P    = xb;                            // overlays dead xb+V (exact fit)
    bf*    attn = Q;                             // Q dead after scores
    bf*    proj = Kb;                            // K dead after scores

    dim3 blk(256);

    // ---- phase 0: precision conversion / weight transposes -----------------
    transpose_cvt<<<dim3(D / 32,   D / 32, 1),  blk, 0, stream>>>(wq, wqT, D, D);
    transpose_cvt<<<dim3(D / 32,   D / 32, 1),  blk, 0, stream>>>(wk, wkT, D, D);
    transpose_cvt<<<dim3(D / 32,   D / 32, 1),  blk, 0, stream>>>(wv, wvT, D, D);
    transpose_cvt<<<dim3(D / 32,   D / 32, 1),  blk, 0, stream>>>(wo, woT, D, D);
    transpose_cvt<<<dim3(DFF / 32, D / 32, 1),  blk, 0, stream>>>(w1, w1T, D, DFF);
    transpose_cvt<<<dim3(D / 32, DFF / 32, 1),  blk, 0, stream>>>(w2, w2T, DFF, D);
    long long nx = (long long)B * S * D;
    cvt_bf16<<<dim3((unsigned)((nx + 255) / 256)), blk, 0, stream>>>(x, xb, nx);

    // ---- phase 1: Q,K,V = x @ W   (M = B*S folded) --------------------------
    dim3 gQKV(D / TN, (B * S) / TM, 1);
    wmma_gemm<0,0,0><<<gQKV, blk, 0, stream>>>(xb, wqT, Q,  nullptr, B*S, D, D, 0,0,0);
    wmma_gemm<0,0,0><<<gQKV, blk, 0, stream>>>(xb, wkT, Kb, nullptr, B*S, D, D, 0,0,0);
    wmma_gemm<0,0,0><<<gQKV, blk, 0, stream>>>(xb, wvT, V,  nullptr, B*S, D, D, 0,0,0);

    // ---- phase 2: Vt = transpose(V) per batch -------------------------------
    transpose_bf16<<<dim3(D / 32, S / 32, B), blk, 0, stream>>>(V, Vt, S, D);

    // ---- phase 3: scores = Q @ K^T (fp32), per batch ------------------------
    wmma_gemm<0,1,0><<<dim3(S / TN, S / TM, B), blk, 0, stream>>>(
        Q, Kb, scores, nullptr, S, S, D,
        (long long)S * D, (long long)S * D, (long long)S * S);

    // ---- phase 4: P = softmax(scores) --------------------------------------
    softmax_rows<<<dim3(B * S), blk, 0, stream>>>(scores, P, S);

    // ---- phase 5: attn = P @ V  (BT = Vt), per batch ------------------------
    wmma_gemm<0,0,0><<<dim3(D / TN, S / TM, B), blk, 0, stream>>>(
        P, Vt, attn, nullptr, S, D, S,
        (long long)S * S, (long long)D * S, (long long)S * D);

    // ---- phase 6: proj = attn @ wo -----------------------------------------
    wmma_gemm<0,0,0><<<gQKV, blk, 0, stream>>>(attn, woT, proj, nullptr, B*S, D, D, 0,0,0);

    // ---- phase 7: hidden = relu(proj @ w1) ----------------------------------
    wmma_gemm<1,0,0><<<dim3(DFF / TN, (B * S) / TM, 1), blk, 0, stream>>>(
        proj, w1T, hidden, nullptr, B*S, DFF, D, 0,0,0);

    // ---- phase 8: out = hidden @ w2 + x  (fp32 out + residual) --------------
    wmma_gemm<0,1,1><<<gQKV, blk, 0, stream>>>(
        hidden, w2T, d_out, x, B*S, D, DFF, 0,0,0);
}
